// CounterPropagationNetwork_80590766342611
// MI455X (gfx1250) — matrix operand order repaired and checked by reading, not compiled, not run on hardware
//
#include <hip/hip_runtime.h>
#include <hip/hip_bf16.h>

// Problem sizes (fixed by reference)
#define B_SZ 16384
#define D_SZ 256
#define H_SZ 4096
#define O_SZ 512

typedef __attribute__((ext_vector_type(16))) _Float16 v16h;
typedef __attribute__((ext_vector_type(8)))  _Float16 v8h;
typedef __attribute__((ext_vector_type(4)))  _Float16 v4h;
typedef __attribute__((ext_vector_type(8)))  float    v8f;
typedef __attribute__((ext_vector_type(4)))  float    v4f;

// ---------------------------------------------------------------------------
// 1) Split x (fp32) into f16 hi + f16 lo residual: x ~= hi + lo, |lo| <= 2^-11|x|
// ---------------------------------------------------------------------------
__global__ __launch_bounds__(256) void cpn_cvt_x(const float* __restrict__ x,
                                                 _Float16* __restrict__ xh,
                                                 _Float16* __restrict__ xl) {
    int i = blockIdx.x * 256 + threadIdx.x;          // one float4 per thread
    v4f v  = ((const v4f*)x)[i];
    v4h hi = __builtin_convertvector(v, v4h);
    v4f bk = __builtin_convertvector(hi, v4f);
    v4h lo = __builtin_convertvector(v - bk, v4h);
    ((v4h*)xh)[i] = hi;
    ((v4h*)xl)[i] = lo;
}

// ---------------------------------------------------------------------------
// 2) Split kohonen weights into f16 hi/lo and compute 0.5*||w_h||^2 (fp32).
//    One wave (32 lanes) per row of 256 elements.
// ---------------------------------------------------------------------------
__global__ __launch_bounds__(256) void cpn_cvt_w(const float* __restrict__ w,
                                                 _Float16* __restrict__ wh,
                                                 _Float16* __restrict__ wl,
                                                 float* __restrict__ hn) {
    const int lane = threadIdx.x & 31;
    const int wv   = threadIdx.x >> 5;
    const int h    = blockIdx.x * 8 + wv;
    const float* row = w + (size_t)h * D_SZ;
    float ss = 0.0f;
#pragma unroll
    for (int j = 0; j < 8; ++j) {
        int d = lane + j * 32;
        float v = row[d];
        ss += v * v;
        _Float16 hi = (_Float16)v;
        _Float16 lo = (_Float16)(v - (float)hi);
        wh[(size_t)h * D_SZ + d] = hi;
        wl[(size_t)h * D_SZ + d] = lo;
    }
#pragma unroll
    for (int off = 16; off > 0; off >>= 1)
        ss += __shfl_xor(ss, off, 32);
    if (lane == 0) hn[h] = 0.5f * ss;
}

// ---------------------------------------------------------------------------
// 3) Main fused GEMM + argmin kernel.
//    Wave tile: 16 rows of x vs all H kohonen rows, N-tiles of 16.
//    score[b,h] = 0.5*||w_h||^2 - x_b . w_h   (same argmin as the distance)
//    Dot product via split-f16: ah*bh + al*bh + ah*bl  (near-fp32 accuracy).
//    Three independent accumulators -> 3 concurrent WMMA chains per wave.
// ---------------------------------------------------------------------------
__global__ __launch_bounds__(256) void cpn_kohonen_winner(
    const _Float16* __restrict__ xh, const _Float16* __restrict__ xl,
    const _Float16* __restrict__ wh, const _Float16* __restrict__ wl,
    const float* __restrict__ hn, int* __restrict__ winner) {
    const int lane = threadIdx.x & 31;
    const int wave = threadIdx.x >> 5;
    const int gw   = blockIdx.x * 8 + wave;   // global wave-tile id
    const int m0   = gw * 16;                 // base row of this wave's tile

    // ---- Load all A fragments (K = 256 -> 8 fragments of K=32), hi and lo.
    // A 16x32 layout (matching B): lane<16 holds M=lane, K = 0..15;
    // lane>=16 holds M=lane-16, K = 16..31.
    const int arow = m0 + (lane & 15);
    const int kph  = (lane >> 4) * 16;        // K phase within fragment
    v16h ah[8], al[8];
#pragma unroll
    for (int f = 0; f < 8; ++f) {
        const _Float16* ph = xh + (size_t)arow * D_SZ + f * 32 + kph;
        const _Float16* pl = xl + (size_t)arow * D_SZ + f * 32 + kph;
        v8h h0 = *(const v8h*)(ph);
        v8h h1 = *(const v8h*)(ph + 8);
        v8h l0 = *(const v8h*)(pl);
        v8h l1 = *(const v8h*)(pl + 8);
        ah[f] = __builtin_shufflevector(h0, h1, 0,1,2,3,4,5,6,7,8,9,10,11,12,13,14,15);
        al[f] = __builtin_shufflevector(l0, l1, 0,1,2,3,4,5,6,7,8,9,10,11,12,13,14,15);
    }

    float minv[8];
    int   mini[8];
#pragma unroll
    for (int r = 0; r < 8; ++r) { minv[r] = __builtin_huge_valf(); mini[r] = 0; }

    // ---- Sweep all H kohonen rows in N-tiles of 16.
    // B 32x16 layout used here (same k order as A): lane holds column n = lane&15, the 16
    // contiguous K values starting at (lane>>4)*16.
    const int ncol  = lane & 15;
    const int koff  = (lane >> 4) << 4;
    for (int n0 = 0; n0 < H_SZ; n0 += 16) {
        const _Float16* bh_base = wh + (size_t)(n0 + ncol) * D_SZ + koff;
        const _Float16* bl_base = wl + (size_t)(n0 + ncol) * D_SZ + koff;
        // Prefetch next tile's B columns into near cache (WGP scope): the
        // whole workgroup re-reads these lines.
        __builtin_prefetch(bh_base + (size_t)16 * D_SZ, 0, 3);
        __builtin_prefetch(bl_base + (size_t)16 * D_SZ, 0, 3);

        v8f acc0 = {};   // ah * bh
        v8f acc1 = {};   // al * bh
        v8f acc2 = {};   // ah * bl
#pragma unroll
        for (int f = 0; f < 8; ++f) {
            v16h bh = *(const v16h*)(bh_base + f * 32);
            v16h bl = *(const v16h*)(bl_base + f * 32);
            acc0 = __builtin_amdgcn_wmma_f32_16x16x32_f16(false, ah[f], false, bh,
                                                          (short)0, acc0, false, false);
            acc1 = __builtin_amdgcn_wmma_f32_16x16x32_f16(false, al[f], false, bh,
                                                          (short)0, acc1, false, false);
            acc2 = __builtin_amdgcn_wmma_f32_16x16x32_f16(false, ah[f], false, bl,
                                                          (short)0, acc2, false, false);
        }

        const int   nidx = n0 + ncol;
        const float hnv  = hn[nidx];
        // C layout: lane holds column nidx; VGPR r holds row (r + 8*(lane>>4)).
#pragma unroll
        for (int r = 0; r < 8; ++r) {
            float s = hnv - (acc0[r] + (acc1[r] + acc2[r]));
            if (s < minv[r]) { minv[r] = s; mini[r] = nidx; }
        }
    }

    // ---- Cross-lane argmin: row m = r + 8*(lane>>4) lives in the 16 lanes of
    // one half-wave; xor masks 1,2,4,8 reduce within that half. Ties -> lowest
    // index (matches jnp.argmin first-min semantics).
#pragma unroll
    for (int r = 0; r < 8; ++r) {
        float v = minv[r];
        int   i = mini[r];
#pragma unroll
        for (int off = 1; off <= 8; off <<= 1) {
            float ov = __shfl_xor(v, off, 32);
            int   oi = __shfl_xor(i, off, 32);
            if (ov < v || (ov == v && oi < i)) { v = ov; i = oi; }
        }
        if ((lane & 15) == 0)
            winner[m0 + r + ((lane >> 4) * 8)] = i;
    }
}

// ---------------------------------------------------------------------------
// 4) Transpose grossberg [O,H] -> gT [H,O] so the gather is coalesced.
// ---------------------------------------------------------------------------
__global__ __launch_bounds__(256) void cpn_transpose_g(const float* __restrict__ g,
                                                       float* __restrict__ gT) {
    __shared__ float tile[32][33];
    const int x0 = blockIdx.x * 32;   // H dimension
    const int y0 = blockIdx.y * 32;   // O dimension
    const int tx = threadIdx.x, ty = threadIdx.y;   // (32, 8)
#pragma unroll
    for (int j = 0; j < 32; j += 8)
        tile[ty + j][tx] = g[(size_t)(y0 + ty + j) * H_SZ + (x0 + tx)];
    __syncthreads();
#pragma unroll
    for (int j = 0; j < 32; j += 8)
        gT[(size_t)(x0 + ty + j) * O_SZ + (y0 + tx)] = tile[tx][ty + j];
}

// ---------------------------------------------------------------------------
// 5) Gather: out[b,:] = gT[winner[b],:]  (one block per row, float4 copies)
// ---------------------------------------------------------------------------
__global__ __launch_bounds__(128) void cpn_gather(const float* __restrict__ gT,
                                                  const int* __restrict__ winner,
                                                  float* __restrict__ out) {
    const int b = blockIdx.x;
    const int h = winner[b];
    const float4* src = (const float4*)(gT + (size_t)h * O_SZ);
    float4*       dst = (float4*)(out + (size_t)b * O_SZ);
    dst[threadIdx.x] = src[threadIdx.x];   // 128 threads * 16B = 2 KB row
}

// ---------------------------------------------------------------------------
extern "C" void kernel_launch(void* const* d_in, const int* in_sizes, int n_in,
                              void* d_out, int out_size, void* d_ws, size_t ws_size,
                              hipStream_t stream) {
    (void)in_sizes; (void)n_in; (void)out_size; (void)ws_size;
    const float* x  = (const float*)d_in[0];   // [B, D]
    const float* kw = (const float*)d_in[1];   // [H, D]
    const float* gw = (const float*)d_in[2];   // [O, H]

    float* outf = (float*)d_out;                         // [B, O] fp32
    int*   widx = (int*)(outf + (size_t)B_SZ * O_SZ);    // [B] int32 (tuple part 2)

    // Workspace layout (bytes)
    char* ws = (char*)d_ws;
    _Float16* xh = (_Float16*)(ws + 0);                  //  8 MiB
    _Float16* xl = (_Float16*)(ws + 8388608);            //  8 MiB
    _Float16* wh = (_Float16*)(ws + 16777216);           //  2 MiB
    _Float16* wl = (_Float16*)(ws + 18874368);           //  2 MiB
    float*    hn = (float*)   (ws + 20971520);           // 16 KiB
    float*    gT = (float*)   (ws + 20987904);           //  8 MiB

    // 1) split x into f16 hi/lo
    cpn_cvt_x<<<(B_SZ * D_SZ / 4) / 256, 256, 0, stream>>>(x, xh, xl);
    // 2) split kohonen weights + 0.5*||w||^2
    cpn_cvt_w<<<H_SZ / 8, 256, 0, stream>>>(kw, wh, wl, hn);
    // 3) transpose grossberg for coalesced gather
    cpn_transpose_g<<<dim3(H_SZ / 32, O_SZ / 32), dim3(32, 8), 0, stream>>>(gw, gT);
    // 4) fused WMMA GEMM + argmin -> winners
    cpn_kohonen_winner<<<B_SZ / (16 * 8), 256, 0, stream>>>(xh, xl, wh, wl, hn, widx);
    // 5) gather grossberg rows -> output
    cpn_gather<<<B_SZ, 128, 0, stream>>>(gT, widx, outf);
}
